// DeepKoopman_80221399155271
// MI455X (gfx1250) — matrix operand order repaired and checked by reading, not compiled, not run on hardware
//
#include <hip/hip_runtime.h>

// ---------------- dims (fixed by the harness reference) ----------------
#define BB 1024
#define SS 512
#define DD 32
#define LL 64
#define HH 64
#define PP 256
#define BS (BB * SS)          // 524288 rows through the main path
#define TPW 4                 // tiles per wave in the main kernel

typedef _Float16 half_t;
typedef __attribute__((ext_vector_type(16))) _Float16 v16h;
typedef __attribute__((ext_vector_type(8)))  _Float16 v8h;
typedef __attribute__((ext_vector_type(8)))  float    v8f;

// ---------------- WMMA wrapper ----------------
static __device__ __forceinline__ v8f wmma_f16(v16h a, v16h b, v8f c) {
  // D = A(16x32 f16) * B(32x16 f16) + C(16x16 f32)
  return __builtin_amdgcn_wmma_f32_16x16x32_f16(
      /*neg_a=*/false, a, /*neg_b=*/false, b,
      /*c_mod=*/(short)0, c, /*reuse_a=*/false, /*reuse_b=*/false);
}

// Wave-local LDS staging sync: DS ops from one wave complete in order at the
// LDS, so we only need compiler-level ordering (free), not s_barrier.
static __device__ __forceinline__ void wave_sync() {
  __builtin_amdgcn_fence(__ATOMIC_RELEASE, "wavefront");
  __builtin_amdgcn_wave_barrier();
  __builtin_amdgcn_fence(__ATOMIC_ACQUIRE, "wavefront");
}

// ---------------- fragment helpers (wave32, CDNA5 layouts) ----------------
// A fragment (16-bit, 16x32): lane = M%16 | (K-half select); per lane two
// contiguous 8-half runs at cols  col0+klo  and  col0+16+klo  (klo = 0 or 8).
static __device__ __forceinline__ v16h load_a_f16(const half_t* S, int ld, int col0) {
  int lane = threadIdx.x & 31;
  int m    = lane & 15;
  int klo  = (lane >> 4) << 3;              // 0 or 8
  const half_t* p0 = S + m * ld + col0 + klo;
  v8h lo = *(const v8h*)(p0);
  v8h hi = *(const v8h*)(p0 + 16);
  return __builtin_shufflevector(lo, hi, 0,1,2,3,4,5,6,7,8,9,10,11,12,13,14,15);
}

// Same A fragment but sourced from row-major f32 with on-the-fly cvt to f16.
static __device__ __forceinline__ v16h load_a_f32(const float* S, long ld, int col0) {
  int lane = threadIdx.x & 31;
  int m    = lane & 15;
  int klo  = (lane >> 4) << 3;
  const float* p0 = S + (long)m * ld + col0 + klo;
  const float* p1 = p0 + 16;
  v16h a;
#pragma unroll
  for (int i = 0; i < 8; ++i) a[i] = (_Float16)p0[i];
#pragma unroll
  for (int i = 0; i < 8; ++i) a[8 + i] = (_Float16)p1[i];
  return a;
}

// B fragment (16-bit, 32x16): lane = N%16 | (K-half select); per lane rows
// kbase+klo+i and kbase+16+klo+i (i = 0..7, klo = 0 or 8) of column nbase+N%16.
static __device__ __forceinline__ v16h load_b_f16(const half_t* S, int ld, int kbase, int nbase) {
  int lane = threadIdx.x & 31;
  int klo  = (lane >> 4) << 3;
  const half_t* p = S + (kbase + klo) * ld + nbase + (lane & 15);
  v16h b;
#pragma unroll
  for (int i = 0; i < 8; ++i) { b[i] = p[i * ld]; b[8 + i] = p[(16 + i) * ld]; }
  return b;
}

// C/D 32-bit 16x16 layout: lane l, vgpr v -> m = v + 8*(l>=16), n = l%16.
static __device__ __forceinline__ void store_c_f16(half_t* S, int ld, int nbase, v8f c) {
  int lane = threadIdx.x & 31;
  int n    = lane & 15;
  int mhi  = (lane >> 4) << 3;
#pragma unroll
  for (int v = 0; v < 8; ++v) S[(mhi + v) * ld + nbase + n] = (_Float16)c[v];
}

static __device__ __forceinline__ void store_c_f32(float* G, long ld, int nbase, v8f c) {
  int lane = threadIdx.x & 31;
  int n    = lane & 15;
  int mhi  = (lane >> 4) << 3;
#pragma unroll
  for (int v = 0; v < 8; ++v) G[(long)(mhi + v) * ld + nbase + n] = c[v];
}

static __device__ __forceinline__ v8f bias_act(v8f c, const float* bias, int nbase, bool relu) {
  int n = threadIdx.x & 15;
  float bv = bias[nbase + n];
#pragma unroll
  for (int i = 0; i < 8; ++i) {
    float t = c[i] + bv;
    c[i] = relu ? fmaxf(t, 0.0f) : t;
  }
  return c;
}

// =====================================================================
// Kernel 1: Koopman operator powers, log-doubling, stored in transposed
// "B-fragment ready" layout:  Kp[p][k][n] = (K^(p+1))[n][k]  (f16).
// Composition in this storage is a plain row-major matmul.
// =====================================================================
__global__ __launch_bounds__(256) void kpow_init(const float* __restrict__ Kw,
                                                 half_t* __restrict__ Kp) {
  int i = blockIdx.x * 256 + threadIdx.x;     // 64*64 = 4096 elements
  if (i < LL * LL) {
    int k = i >> 6, n = i & 63;
    Kp[i] = (_Float16)Kw[n * LL + k];
  }
}

__global__ __launch_bounds__(128) void kpow_round(half_t* __restrict__ Kp, int step) {
  // block o produces power (step + o + 1) = B_step @ B_{o+1}  (row-major)
  int o = blockIdx.x;
  const half_t* A   = Kp + (long)(step - 1) * (LL * LL);
  const half_t* Bm  = Kp + (long)o * (LL * LL);
  half_t*       Dst = Kp + (long)(step + o) * (LL * LL);

  int wave = threadIdx.x >> 5;                // 4 waves: row tiles of 16
  const half_t* Arow = A + wave * 16 * LL;
  v16h a0 = load_a_f16(Arow, LL, 0);
  v16h a1 = load_a_f16(Arow, LL, 32);
#pragma unroll
  for (int nt = 0; nt < 4; ++nt) {
    v8f c = {};
    c = wmma_f16(a0, load_b_f16(Bm, LL, 0,  nt * 16), c);
    c = wmma_f16(a1, load_b_f16(Bm, LL, 32, nt * 16), c);
    store_c_f16(Dst + wave * 16 * LL, LL, nt * 16, c);
  }
}

// =====================================================================
// Kernel 2: fused main path. Per 16-row tile (one wave, TPW tiles/wave):
//   h = relu(x@W1^T+b1); z = h@W2^T+b2; zd = z@K^T;
//   x_rec = dec(z); x_dyn = dec(zd); z, zd also written out.
// =====================================================================
__global__ __launch_bounds__(256) void koop_main(
    const float* __restrict__ x,
    const float* __restrict__ ew1, const float* __restrict__ eb1,
    const float* __restrict__ ew2, const float* __restrict__ eb2,
    const float* __restrict__ dw1, const float* __restrict__ db1,
    const float* __restrict__ dw2, const float* __restrict__ db2,
    const float* __restrict__ Kw,
    float* __restrict__ x_rec, float* __restrict__ x_dyn,
    float* __restrict__ z_out, float* __restrict__ zd_out) {
  // Weights in B-matrix layout  W[k][n] = w[n][k]  (f16).
  __shared__ __attribute__((aligned(32))) half_t W1[DD * HH];   //  4 KB
  __shared__ __attribute__((aligned(32))) half_t W2[HH * LL];   //  8 KB
  __shared__ __attribute__((aligned(32))) half_t KB[LL * LL];   //  8 KB
  __shared__ __attribute__((aligned(32))) half_t D1[LL * HH];   //  8 KB
  __shared__ __attribute__((aligned(32))) half_t D2[HH * DD];   //  4 KB
  __shared__ float BE1[HH], BE2[LL], BD1[HH], BD2[DD];
  __shared__ __attribute__((aligned(32))) half_t stage[8][16 * 64];  // 16 KB

  int t = threadIdx.x;
  for (int i = t; i < DD * HH; i += 256) { int d = i >> 6, h = i & 63; W1[i] = (_Float16)ew1[h * DD + d]; }
  for (int i = t; i < HH * LL; i += 256) { int h = i >> 6, l = i & 63; W2[i] = (_Float16)ew2[l * HH + h]; }
  for (int i = t; i < LL * LL; i += 256) { int k = i >> 6, n = i & 63; KB[i] = (_Float16)Kw[n * LL + k]; }
  for (int i = t; i < LL * HH; i += 256) { int l = i >> 6, h = i & 63; D1[i] = (_Float16)dw1[h * LL + l]; }
  for (int i = t; i < HH * DD; i += 256) { int h = i >> 5, d = i & 31; D2[i] = (_Float16)dw2[d * HH + h]; }
  if (t < HH) { BE1[t] = eb1[t]; BE2[t] = eb2[t]; BD1[t] = db1[t]; }
  if (t < DD) { BD2[t] = db2[t]; }
  __syncthreads();   // weights ready for all waves (only cross-wave barrier)

  int     wave = t >> 5;
  half_t* buf  = stage[wave];
  long    base = ((long)blockIdx.x * 8 + wave) * TPW;   // first tile of this wave

#pragma unroll 1
  for (int it = 0; it < TPW; ++it) {
    long r0 = (base + it) * 16;

    // prefetch next tile's x rows while this tile computes
    if (it + 1 < TPW) __builtin_prefetch(x + (r0 + 16) * DD, 0, 1);

    // ---- stage 1: h = relu(x @ W1^T + b1) ----
    v16h ax = load_a_f32(x + r0 * DD, DD, 0);
#pragma unroll
    for (int nt = 0; nt < 4; ++nt) {
      v8f c = {};
      c = wmma_f16(ax, load_b_f16(W1, HH, 0, nt * 16), c);
      c = bias_act(c, BE1, nt * 16, true);
      store_c_f16(buf, HH, nt * 16, c);
    }
    wave_sync();
    v16h ah0 = load_a_f16(buf, HH, 0), ah1 = load_a_f16(buf, HH, 32);
    wave_sync();

    // ---- stage 2: z = h @ W2^T + b2  (write f32 out + f16 stage) ----
#pragma unroll
    for (int nt = 0; nt < 4; ++nt) {
      v8f c = {};
      c = wmma_f16(ah0, load_b_f16(W2, LL, 0,  nt * 16), c);
      c = wmma_f16(ah1, load_b_f16(W2, LL, 32, nt * 16), c);
      c = bias_act(c, BE2, nt * 16, false);
      store_c_f32(z_out + r0 * LL, LL, nt * 16, c);
      store_c_f16(buf, LL, nt * 16, c);
    }
    wave_sync();
    v16h az0 = load_a_f16(buf, LL, 0), az1 = load_a_f16(buf, LL, 32);
    wave_sync();

    // ---- stage 3: z_dyn = z @ K^T ----
#pragma unroll
    for (int nt = 0; nt < 4; ++nt) {
      v8f c = {};
      c = wmma_f16(az0, load_b_f16(KB, LL, 0,  nt * 16), c);
      c = wmma_f16(az1, load_b_f16(KB, LL, 32, nt * 16), c);
      store_c_f32(zd_out + r0 * LL, LL, nt * 16, c);
      store_c_f16(buf, LL, nt * 16, c);
    }
    wave_sync();
    v16h ad0 = load_a_f16(buf, LL, 0), ad1 = load_a_f16(buf, LL, 32);
    wave_sync();

    // ---- stage 4: hr = relu(z @ D1^T + bd1) ----
#pragma unroll
    for (int nt = 0; nt < 4; ++nt) {
      v8f c = {};
      c = wmma_f16(az0, load_b_f16(D1, HH, 0,  nt * 16), c);
      c = wmma_f16(az1, load_b_f16(D1, HH, 32, nt * 16), c);
      c = bias_act(c, BD1, nt * 16, true);
      store_c_f16(buf, HH, nt * 16, c);
    }
    wave_sync();
    v16h ar0 = load_a_f16(buf, HH, 0), ar1 = load_a_f16(buf, HH, 32);
    wave_sync();

    // ---- stage 5: x_rec = hr @ D2^T + bd2 ----
#pragma unroll
    for (int nt = 0; nt < 2; ++nt) {
      v8f c = {};
      c = wmma_f16(ar0, load_b_f16(D2, DD, 0,  nt * 16), c);
      c = wmma_f16(ar1, load_b_f16(D2, DD, 32, nt * 16), c);
      c = bias_act(c, BD2, nt * 16, false);
      store_c_f32(x_rec + r0 * DD, DD, nt * 16, c);
    }

    // ---- stage 6: hd = relu(z_dyn @ D1^T + bd1) ----
#pragma unroll
    for (int nt = 0; nt < 4; ++nt) {
      v8f c = {};
      c = wmma_f16(ad0, load_b_f16(D1, HH, 0,  nt * 16), c);
      c = wmma_f16(ad1, load_b_f16(D1, HH, 32, nt * 16), c);
      c = bias_act(c, BD1, nt * 16, true);
      store_c_f16(buf, HH, nt * 16, c);
    }
    wave_sync();
    v16h aq0 = load_a_f16(buf, HH, 0), aq1 = load_a_f16(buf, HH, 32);
    wave_sync();

    // ---- stage 7: x_dyn = hd @ D2^T + bd2 ----
#pragma unroll
    for (int nt = 0; nt < 2; ++nt) {
      v8f c = {};
      c = wmma_f16(aq0, load_b_f16(D2, DD, 0,  nt * 16), c);
      c = wmma_f16(aq1, load_b_f16(D2, DD, 32, nt * 16), c);
      c = bias_act(c, BD2, nt * 16, false);
      store_c_f32(x_dyn + r0 * DD, DD, nt * 16, c);
    }
    wave_sync();   // buf reuse across iterations
  }
}

// =====================================================================
// Kernel 3: prediction path, fully parallel over (b-tile, p):
//   z_p = z0 @ (K^(p+1))^T  (B-operand straight from precomputed Kp[p]),
//   x_pred[:, p, :] = dec(z_p).   4 p-values per wave per block.
// =====================================================================
__global__ __launch_bounds__(256) void koop_pred(
    const float* __restrict__ z_full, const half_t* __restrict__ Kp,
    const float* __restrict__ dw1, const float* __restrict__ db1,
    const float* __restrict__ dw2, const float* __restrict__ db2,
    float* __restrict__ x_pred) {
  __shared__ __attribute__((aligned(32))) half_t D1[LL * HH];   // 8 KB
  __shared__ __attribute__((aligned(32))) half_t D2[HH * DD];   // 4 KB
  __shared__ float BD1[HH], BD2[DD];
  __shared__ __attribute__((aligned(32))) half_t z0s[16 * LL];  // 2 KB
  __shared__ __attribute__((aligned(32))) half_t stage[8][16 * 64];

  int t = threadIdx.x;
  for (int i = t; i < LL * HH; i += 256) { int l = i >> 6, h = i & 63; D1[i] = (_Float16)dw1[h * LL + l]; }
  for (int i = t; i < HH * DD; i += 256) { int h = i >> 5, d = i & 31; D2[i] = (_Float16)dw2[d * HH + h]; }
  if (t < HH) BD1[t] = db1[t];
  if (t < DD) BD2[t] = db2[t];

  int btile = blockIdx.x;                       // 64 tiles over B
  for (int i = t; i < 16 * LL; i += 256) {      // z0 = z[:, S-1, :] tile -> f16
    int m = i >> 6, col = i & 63;
    long b = (long)btile * 16 + m;
    z0s[i] = (_Float16)z_full[(b * SS + (SS - 1)) * LL + col];
  }
  __syncthreads();

  int     wave = t >> 5;
  half_t* buf  = stage[wave];
  v16h az0 = load_a_f16(z0s, LL, 0), az1 = load_a_f16(z0s, LL, 32);

#pragma unroll 1
  for (int i = 0; i < 4; ++i) {
    int p = blockIdx.y * 32 + i * 8 + wave;     // 0..255

    const half_t* Bp = Kp + (long)p * (LL * LL);
    // z_p tile
#pragma unroll
    for (int nt = 0; nt < 4; ++nt) {
      v8f c = {};
      c = wmma_f16(az0, load_b_f16(Bp, LL, 0,  nt * 16), c);
      c = wmma_f16(az1, load_b_f16(Bp, LL, 32, nt * 16), c);
      store_c_f16(buf, LL, nt * 16, c);
    }
    wave_sync();
    v16h ap0 = load_a_f16(buf, LL, 0), ap1 = load_a_f16(buf, LL, 32);
    wave_sync();

    // hp = relu(z_p @ D1^T + bd1)
#pragma unroll
    for (int nt = 0; nt < 4; ++nt) {
      v8f c = {};
      c = wmma_f16(ap0, load_b_f16(D1, HH, 0,  nt * 16), c);
      c = wmma_f16(ap1, load_b_f16(D1, HH, 32, nt * 16), c);
      c = bias_act(c, BD1, nt * 16, true);
      store_c_f16(buf, HH, nt * 16, c);
    }
    wave_sync();
    v16h ah0 = load_a_f16(buf, HH, 0), ah1 = load_a_f16(buf, HH, 32);
    wave_sync();

    // x_pred[b0..b0+15, p, :]   (row stride P*D between b rows)
    float* out = x_pred + ((long)btile * 16 * PP + p) * DD;
#pragma unroll
    for (int nt = 0; nt < 2; ++nt) {
      v8f c = {};
      c = wmma_f16(ah0, load_b_f16(D2, DD, 0,  nt * 16), c);
      c = wmma_f16(ah1, load_b_f16(D2, DD, 32, nt * 16), c);
      c = bias_act(c, BD2, nt * 16, false);
      store_c_f32(out, (long)PP * DD, nt * 16, c);
    }
  }
}

// =====================================================================
extern "C" void kernel_launch(void* const* d_in, const int* in_sizes, int n_in,
                              void* d_out, int out_size, void* d_ws, size_t ws_size,
                              hipStream_t stream) {
  (void)in_sizes; (void)n_in; (void)out_size; (void)ws_size;
  const float* x   = (const float*)d_in[0];
  const float* ew1 = (const float*)d_in[1];
  const float* eb1 = (const float*)d_in[2];
  const float* ew2 = (const float*)d_in[3];
  const float* eb2 = (const float*)d_in[4];
  const float* dw1 = (const float*)d_in[5];
  const float* db1 = (const float*)d_in[6];
  const float* dw2 = (const float*)d_in[7];
  const float* db2 = (const float*)d_in[8];
  const float* Kw  = (const float*)d_in[9];
  // d_in[10] = pred_len (==PP, fixed by the harness)

  float* out    = (float*)d_out;
  float* x_rec  = out;
  float* x_dyn  = out + (long)BS * DD;                       // 16777216
  float* x_pred = out + 2L * BS * DD;                        // 33554432
  float* z_o    = out + 2L * BS * DD + (long)BB * PP * DD;   // 41943040
  float* zd_o   = z_o + (long)BS * LL;                       // 75497472

  half_t* Kp = (half_t*)d_ws;                                // 256 * 64*64 f16 = 2 MB

  // K^1 ... K^256 via log-doubling (8 tiny rounds, growing parallelism)
  kpow_init<<<16, 256, 0, stream>>>(Kw, Kp);
  for (int step = 1; step < PP; step <<= 1) {
    int count = (PP - step) < step ? (PP - step) : step;
    kpow_round<<<count, 128, 0, stream>>>(Kp, step);
  }

  // Fused main path: 32768 tiles of 16 rows, 8 waves / block, 4 tiles / wave
  koop_main<<<BS / (16 * 8 * TPW), 256, 0, stream>>>(x, ew1, eb1, ew2, eb2,
                                                     dw1, db1, dw2, db2, Kw,
                                                     x_rec, x_dyn, z_o, zd_o);

  // Prediction: 64 b-tiles x 8 p-groups, 8 waves / block, 4 p / wave
  koop_pred<<<dim3(64, 8), 256, 0, stream>>>(z_o, Kp, dw1, db1, dw2, db2, x_pred);
}
